// FasterGRUTree_67130338836778
// MI455X (gfx1250) — matrix operand order repaired and checked by reading, not compiled, not run on hardware
//
#include <hip/hip_runtime.h>
#include <hip/hip_bf16.h>

// ---------------------------------------------------------------------------
// FasterGRUTree on MI455X (gfx1250), all GEMMs via v_wmma_f32_16x16x32_f16.
// B=20000 trees, C=8 children, WORD=300, TAG=50, MEM=150, gates 3*MEM=450.
//
// Pipeline:
//  1) pad_weight x4 : f32 weights -> zero-padded f16 (K padded to mult. of 32)
//  2) pack_xleaf    : concat(emb,tag) -> f16 [160000 x 352]
//  3) leaf_gi_kernel: per 32 rows: leaf GEMM (K=352) -> GRU epilogue ->
//                     gi GEMM (K=256, A from LDS) -> gi_pre f16 [160000 x 464]
//                     (hoists ALL x-side work out of the sequential scan)
//  4) seq_kernel    : per 32 trees: recurrent weights staged once into CDNA5's
//                     big WGP LDS (148.5 KB resident), then 8 sequential gh
//                     GEMMs (K=160, h + W both in LDS) + GRU epilogue reading
//                     gi_pre; then node GRU (LDS reloaded with w_hh_node).
// ---------------------------------------------------------------------------

typedef __attribute__((ext_vector_type(16))) _Float16 v16h;
typedef __attribute__((ext_vector_type(8)))  _Float16 v8h;
typedef __attribute__((ext_vector_type(8)))  float    v8f;

#define NB     20000
#define NC     8
#define NROWS  (NB*NC)        // 160000 leaf rows
#define KLEAF  352            // padded 350
#define KCH    256            // padded 250
#define KHH    160            // padded 150
#define NPAD   464            // padded 450
#define NTILES 29

// seq_kernel dynamic LDS partition (bytes)
#define SW_BYTES   (NPAD * KHH * 2)                 // 148480: resident weights
#define SGH_BYTES  (32 * NPAD * 2)                  //  29696: gate staging
#define SHF_BYTES  (32 * 150 * 4)                   //  19200: h carry (f32)
#define SHH_BYTES  (32 * KHH * 2)                   //  10240: h for WMMA (f16)
#define SEQ_LDS    (SW_BYTES + SGH_BYTES + SHF_BYTES + SHH_BYTES)  // 207616

__device__ __forceinline__ v8f wmma_f16(v16h a, v16h b, v8f c) {
  return __builtin_amdgcn_wmma_f32_16x16x32_f16(
      false, a, false, b, (short)0, c, false, false);
}

// A-fragment load per CDNA5 16-bit A layout; p = rowBase + kBase + (lane>>4)*8
__device__ __forceinline__ v16h load_a16(const _Float16* p) {
  v8h lo = *(const v8h*)(p);        // K = base .. base+7
  v8h hi = *(const v8h*)(p + 16);   // K = base+16 .. base+23
  return __builtin_shufflevector(lo, hi, 0,1,2,3,4,5,6,7,8,9,10,11,12,13,14,15);
}

__device__ __forceinline__ float sigm(float x) {
  return 1.0f / (1.0f + __expf(-x));
}

// ---------------------------------------------------------------------------
// Packing kernels
// ---------------------------------------------------------------------------
__global__ void pad_weight(const float* __restrict__ src, _Float16* __restrict__ dst,
                           int cols_src, int cols_pad, long total) {
  long idx = (long)blockIdx.x * blockDim.x + threadIdx.x;
  if (idx >= total) return;
  int n = (int)(idx / cols_pad);
  int k = (int)(idx - (long)n * cols_pad);
  float v = (n < 450 && k < cols_src) ? src[(long)n * cols_src + k] : 0.0f;
  dst[idx] = (_Float16)v;
}

__global__ void pack_xleaf(const float* __restrict__ embs, const float* __restrict__ tags,
                           _Float16* __restrict__ xleaf, long total) {
  long idx = (long)blockIdx.x * blockDim.x + threadIdx.x;
  if (idx >= total) return;
  long r = idx / KLEAF;
  int  k = (int)(idx - r * KLEAF);
  float v = 0.0f;
  if (k < 300)      v = embs[r * 300 + k];
  else if (k < 350) v = tags[r * 50 + (k - 300)];
  xleaf[idx] = (_Float16)v;
}

// ---------------------------------------------------------------------------
// Fused leaf + gi kernel. Block = 160 threads (5 waves), 32 rows (2 M-tiles).
// ---------------------------------------------------------------------------
__global__ __launch_bounds__(160) void leaf_gi_kernel(
    const _Float16* __restrict__ xleaf, const _Float16* __restrict__ wleaf,
    const _Float16* __restrict__ wihc,
    const float* __restrict__ b_ih_l, const float* __restrict__ b_hh_l,
    const float* __restrict__ tags, const float* __restrict__ tp,
    _Float16* __restrict__ gipre) {
  __shared__ __attribute__((aligned(16))) _Float16 sG[32 * NPAD];  // 29.0 KB
  __shared__ __attribute__((aligned(16))) _Float16 sX[32 * KCH];   // 16.0 KB

  const int tid  = threadIdx.x;
  const int lane = tid & 31;
  const int wave = tid >> 5;
  const int m16  = lane & 15;
  const int lhi  = lane >> 4;
  const long r0  = (long)blockIdx.x * 32;

  // stage static part of x_child rows into sX (cols 150..255)
  for (int idx = tid; idx < 32 * 106; idx += 160) {
    int row = idx / 106;
    int k   = 150 + (idx - row * 106);
    long r  = r0 + row;
    float v = 0.0f;
    if (k < 200)      v = tags[r * 50 + (k - 150)];
    else if (k < 250) v = tp[(r >> 3) * 50 + (k - 200)];
    sX[row * KCH + k] = (_Float16)v;
  }

  // ---- GEMM1: gi_leaf = x_leaf @ W_ih_leaf^T  (A held in registers)
  v16h a0[KLEAF / 32], a1[KLEAF / 32];
  {
    const _Float16* ar0 = xleaf + (r0 + m16) * KLEAF + lhi * 8;
    const _Float16* ar1 = xleaf + (r0 + 16 + m16) * KLEAF + lhi * 8;
#pragma unroll
    for (int kc = 0; kc < KLEAF / 32; ++kc) {
      a0[kc] = load_a16(ar0 + kc * 32);
      a1[kc] = load_a16(ar1 + kc * 32);
    }
  }
  for (int nt = wave; nt < NTILES; nt += 5) {
    v8f acc0 = {}, acc1 = {};
    const _Float16* wrow = wleaf + (long)(nt * 16 + m16) * KLEAF + lhi * 8;
#pragma unroll
    for (int kc = 0; kc < KLEAF / 32; ++kc) {
      v16h b = load_a16(wrow + kc * 32);
      acc0 = wmma_f16(a0[kc], b, acc0);
      acc1 = wmma_f16(a1[kc], b, acc1);
    }
#pragma unroll
    for (int g = 0; g < 8; ++g) {
      sG[(g + lhi * 8) * NPAD + nt * 16 + m16]      = (_Float16)acc0[g];
      sG[(16 + g + lhi * 8) * NPAD + nt * 16 + m16] = (_Float16)acc1[g];
    }
  }
  __syncthreads();

  // ---- leaf GRU epilogue (h0==0 => gh = b_hh); h -> sX[:,0:150]
  for (int idx = tid; idx < 32 * 150; idx += 160) {
    int row = idx / 150, j = idx - row * 150;
    float gr = (float)sG[row * NPAD + j]       + b_ih_l[j]       + b_hh_l[j];
    float gz = (float)sG[row * NPAD + 150 + j] + b_ih_l[150 + j] + b_hh_l[150 + j];
    float gn = (float)sG[row * NPAD + 300 + j] + b_ih_l[300 + j];
    float r = sigm(gr);
    float z = sigm(gz);
    float n = tanhf(gn + r * b_hh_l[300 + j]);
    sX[row * KCH + j] = (_Float16)((1.0f - z) * n);
  }
  __syncthreads();

  // ---- GEMM2: gi_pre = x_child @ W_ih_child^T  (A from LDS, D -> global f16)
  {
    const _Float16* x0 = &sX[m16 * KCH + lhi * 8];
    const _Float16* x1 = &sX[(16 + m16) * KCH + lhi * 8];
    for (int nt = wave; nt < NTILES; nt += 5) {
      v8f acc0 = {}, acc1 = {};
      const _Float16* wi = wihc + (long)(nt * 16 + m16) * KCH + lhi * 8;
#pragma unroll
      for (int kc = 0; kc < KCH / 32; ++kc) {
        v16h b = load_a16(wi + kc * 32);
        acc0 = wmma_f16(load_a16(x0 + kc * 32), b, acc0);
        acc1 = wmma_f16(load_a16(x1 + kc * 32), b, acc1);
      }
#pragma unroll
      for (int g = 0; g < 8; ++g) {
        gipre[(size_t)(r0 + g + lhi * 8) * NPAD + nt * 16 + m16]      = (_Float16)acc0[g];
        gipre[(size_t)(r0 + 16 + g + lhi * 8) * NPAD + nt * 16 + m16] = (_Float16)acc1[g];
      }
    }
  }
}

// ---------------------------------------------------------------------------
// Sequential kernel: 32 trees per block; only the gh GEMM (K=160) is on the
// critical path, and BOTH its operands live in LDS: h (updated in place) and
// the whole padded w_hh_child (148.5 KB, staged once -- CDNA5's 320 KB WGP
// LDS makes the recurrent weights on-chip-resident). Node GRU reuses the
// weight buffer with w_hh_node.
// ---------------------------------------------------------------------------
__global__ __launch_bounds__(160) void seq_kernel(
    const _Float16* __restrict__ gipre,
    const _Float16* __restrict__ whhc, const _Float16* __restrict__ wnode,
    const float* __restrict__ b_ih_c, const float* __restrict__ b_hh_c,
    const float* __restrict__ b_ih_n, const float* __restrict__ b_hh_n,
    float* __restrict__ out) {
  extern __shared__ __attribute__((aligned(16))) char smem[];
  _Float16* sW  = (_Float16*)smem;                                        // 464x160 f16
  _Float16* sGH = (_Float16*)(smem + SW_BYTES);                           // 32x464 f16
  float*    sHf = (float*)   (smem + SW_BYTES + SGH_BYTES);               // 32x150 f32
  _Float16* sHh = (_Float16*)(smem + SW_BYTES + SGH_BYTES + SHF_BYTES);   // 32x160 f16

  const int tid  = threadIdx.x;
  const int lane = tid & 31;
  const int wave = tid >> 5;
  const int m16  = lane & 15;
  const int lhi  = lane >> 4;
  const long t0  = (long)blockIdx.x * 32;

  // stage recurrent weights on-chip (16B vector copies), zero h
  for (int i = tid; i < SW_BYTES / 16; i += 160)
    ((uint4*)sW)[i] = ((const uint4*)whhc)[i];
  for (int i = tid; i < 32 * 150; i += 160) sHf[i] = 0.0f;
  for (int i = tid; i < 32 * KHH; i += 160) sHh[i] = (_Float16)0.0f;
  __syncthreads();

  const _Float16* h0 = &sHh[m16 * KHH + lhi * 8];
  const _Float16* h1 = &sHh[(16 + m16) * KHH + lhi * 8];

  for (int step = 0; step < NC; ++step) {
    for (int nt = wave; nt < NTILES; nt += 5) {
      v8f acc0 = {}, acc1 = {};
      const _Float16* wh = sW + (nt * 16 + m16) * KHH + lhi * 8;
#pragma unroll
      for (int kc = 0; kc < KHH / 32; ++kc) {
        v16h b = load_a16(wh + kc * 32);
        acc0 = wmma_f16(load_a16(h0 + kc * 32), b, acc0);
        acc1 = wmma_f16(load_a16(h1 + kc * 32), b, acc1);
      }
#pragma unroll
      for (int g = 0; g < 8; ++g) {
        sGH[(g + lhi * 8) * NPAD + nt * 16 + m16]      = (_Float16)acc0[g];
        sGH[(16 + g + lhi * 8) * NPAD + nt * 16 + m16] = (_Float16)acc1[g];
      }
    }
    __syncthreads();

    for (int idx = tid; idx < 32 * 150; idx += 160) {
      int row = idx / 150, j = idx - row * 150;
      const _Float16* gp = gipre + ((size_t)(t0 + row) * NC + step) * NPAD;
      float gr  = (float)gp[j]       + b_ih_c[j]
                + (float)sGH[row * NPAD + j]       + b_hh_c[j];
      float gz  = (float)gp[150 + j] + b_ih_c[150 + j]
                + (float)sGH[row * NPAD + 150 + j] + b_hh_c[150 + j];
      float gin = (float)gp[300 + j] + b_ih_c[300 + j];
      float ghn = (float)sGH[row * NPAD + 300 + j] + b_hh_c[300 + j];
      float r = sigm(gr);
      float z = sigm(gz);
      float n = tanhf(gin + r * ghn);
      float hold = sHf[row * 150 + j];
      float hnew = (1.0f - z) * n + z * hold;
      sHf[row * 150 + j] = hnew;
      sHh[row * KHH + j] = (_Float16)hnew;
    }
    __syncthreads();
  }

  // ---- node GRU: reload LDS weights with w_hh_node, then
  //      gi = b_ih_node (x==0); gh = h @ w_hh_node^T + b_hh_node
  for (int i = tid; i < SW_BYTES / 16; i += 160)
    ((uint4*)sW)[i] = ((const uint4*)wnode)[i];
  __syncthreads();

  for (int nt = wave; nt < NTILES; nt += 5) {
    v8f acc0 = {}, acc1 = {};
    const _Float16* wn = sW + (nt * 16 + m16) * KHH + lhi * 8;
#pragma unroll
    for (int kc = 0; kc < KHH / 32; ++kc) {
      v16h b = load_a16(wn + kc * 32);
      acc0 = wmma_f16(load_a16(h0 + kc * 32), b, acc0);
      acc1 = wmma_f16(load_a16(h1 + kc * 32), b, acc1);
    }
#pragma unroll
    for (int g = 0; g < 8; ++g) {
      sGH[(g + lhi * 8) * NPAD + nt * 16 + m16]      = (_Float16)acc0[g];
      sGH[(16 + g + lhi * 8) * NPAD + nt * 16 + m16] = (_Float16)acc1[g];
    }
  }
  __syncthreads();

  for (int idx = tid; idx < 32 * 150; idx += 160) {
    int row = idx / 150, j = idx - row * 150;
    float gr = b_ih_n[j]       + (float)sGH[row * NPAD + j]       + b_hh_n[j];
    float gz = b_ih_n[150 + j] + (float)sGH[row * NPAD + 150 + j] + b_hh_n[150 + j];
    float r = sigm(gr);
    float z = sigm(gz);
    float n = tanhf(b_ih_n[300 + j] + r * ((float)sGH[row * NPAD + 300 + j] + b_hh_n[300 + j]));
    float hold = sHf[row * 150 + j];
    out[(t0 + row) * 150 + j] = (1.0f - z) * n + z * hold;
  }
}

// ---------------------------------------------------------------------------
// Launch
// ---------------------------------------------------------------------------
extern "C" void kernel_launch(void* const* d_in, const int* in_sizes, int n_in,
                              void* d_out, int out_size, void* d_ws, size_t ws_size,
                              hipStream_t stream) {
  (void)in_sizes; (void)n_in; (void)out_size; (void)ws_size;

  const float* embs       = (const float*)d_in[0];
  const float* tags       = (const float*)d_in[1];
  const float* tag_parent = (const float*)d_in[2];
  const float* w_ih_leaf  = (const float*)d_in[3];
  // d_in[4] w_hh_leaf: matmul dead (h0==0); only b_hh_leaf matters.
  const float* b_ih_leaf  = (const float*)d_in[5];
  const float* b_hh_leaf  = (const float*)d_in[6];
  const float* w_ih_child = (const float*)d_in[7];
  const float* w_hh_child = (const float*)d_in[8];
  const float* b_ih_child = (const float*)d_in[9];
  const float* b_hh_child = (const float*)d_in[10];
  // d_in[11] w_ih_node: dead (x==0).
  const float* w_hh_node  = (const float*)d_in[12];
  const float* b_ih_node  = (const float*)d_in[13];
  const float* b_hh_node  = (const float*)d_in[14];
  float* out = (float*)d_out;

  // Workspace (~262 MB, all segments 256B-aligned by construction).
  char* ws = (char*)d_ws;
  _Float16* gipre = (_Float16*)ws; ws += (size_t)NROWS * NPAD  * 2;  // 148.48 MB
  _Float16* xleaf = (_Float16*)ws; ws += (size_t)NROWS * KLEAF * 2;  // 112.64 MB
  _Float16* wleaf = (_Float16*)ws; ws += (size_t)NPAD  * KLEAF * 2;
  _Float16* wihc  = (_Float16*)ws; ws += (size_t)NPAD  * KCH   * 2;
  _Float16* whhc  = (_Float16*)ws; ws += (size_t)NPAD  * KHH   * 2;
  _Float16* wnode = (_Float16*)ws; ws += (size_t)NPAD  * KHH   * 2;

  const int T = 256;
  long n;

  n = (long)NPAD * KLEAF;
  pad_weight<<<(unsigned)((n + T - 1) / T), T, 0, stream>>>(w_ih_leaf, wleaf, 350, KLEAF, n);
  n = (long)NPAD * KCH;
  pad_weight<<<(unsigned)((n + T - 1) / T), T, 0, stream>>>(w_ih_child, wihc, 250, KCH, n);
  n = (long)NPAD * KHH;
  pad_weight<<<(unsigned)((n + T - 1) / T), T, 0, stream>>>(w_hh_child, whhc, 150, KHH, n);
  pad_weight<<<(unsigned)((n + T - 1) / T), T, 0, stream>>>(w_hh_node, wnode, 150, KHH, n);

  n = (long)NROWS * KLEAF;
  pack_xleaf<<<(unsigned)((n + T - 1) / T), T, 0, stream>>>(embs, tags, xleaf, n);

  leaf_gi_kernel<<<NROWS / 32, 160, 0, stream>>>(xleaf, wleaf, wihc,
                                                 b_ih_leaf, b_hh_leaf,
                                                 tags, tag_parent, gipre);

  seq_kernel<<<NB / 32, 160, SEQ_LDS, stream>>>(gipre, whhc, wnode,
                                                b_ih_child, b_hh_child,
                                                b_ih_node, b_hh_node, out);
}
